// TransformerEncoder_77816217468969
// MI455X (gfx1250) — matrix operand order repaired and checked by reading, not compiled, not run on hardware
//
#include <hip/hip_runtime.h>
#include <hip/hip_bf16.h>
#include <math.h>
#include <stdint.h>

// ---------------- geometry (compile-time) ----------------
#define Dm   1024
#define Hh   16
#define HDd  64
#define FFd  4096
#define Bb   8
#define Ntok 1024
#define Mtot (Bb * Ntok)   // 8192
#define KSTEP 64

typedef __attribute__((ext_vector_type(16))) _Float16 v16h;
typedef __attribute__((ext_vector_type(8)))  float    v8f;

__device__ __forceinline__ v8f wmma_f16(v16h a, v16h b, v8f c) {
  // D = A(16x32 f16) * B(32x16 f16) + C(16x16 f32)
  return __builtin_amdgcn_wmma_f32_16x16x32_f16(false, a, false, b, (short)0, c, false, false);
}

// load a v16h fragment from two contiguous 16-byte chunks
__device__ __forceinline__ v16h ld2x8(const _Float16* p0, const _Float16* p1) {
  union { uint4 u[2]; v16h v; } t;
  t.u[0] = *(const uint4*)p0;
  t.u[1] = *(const uint4*)p1;
  return t.v;
}

// CDNA5 async global -> LDS copy (16B per lane), tracked by ASYNCcnt
__device__ __forceinline__ void async_ld_b128(unsigned lds_off, const _Float16* gptr) {
  asm volatile("global_load_async_to_lds_b128 %0, %1, off"
               :: "v"(lds_off), "v"((unsigned long long)(uintptr_t)gptr)
               : "memory");
}
__device__ __forceinline__ void wait_async0() {
  asm volatile("s_wait_asynccnt 0" ::: "memory");
}

// ---------------- fp32 -> f16 weight convert ----------------
__global__ __launch_bounds__(256) void cvt_f16_kernel(const float* __restrict__ in,
                                                      _Float16* __restrict__ out, int n) {
  int i = blockIdx.x * 256 + threadIdx.x;
  if (i < n) out[i] = (_Float16)in[i];
}

// ---------------- layernorm: one block per row of D=1024 ----------------
__device__ __forceinline__ float blk_sum(float s, float* sm) {
  #pragma unroll
  for (int off = 16; off > 0; off >>= 1) s += __shfl_xor(s, off);
  int lane = threadIdx.x & 31, wv = threadIdx.x >> 5;
  __syncthreads();
  if (lane == 0) sm[wv] = s;
  __syncthreads();
  float t = 0.f;
  #pragma unroll
  for (int i = 0; i < 8; i++) t += sm[i];
  return t;
}

__global__ __launch_bounds__(256) void ln_kernel(const float* __restrict__ x,
                                                 const float* __restrict__ g,
                                                 const float* __restrict__ bta,
                                                 _Float16* __restrict__ out) {
  __shared__ float sm[8];
  int row = blockIdx.x;
  const float* xr = x + (size_t)row * Dm;
  float v[4]; float s = 0.f;
  #pragma unroll
  for (int i = 0; i < 4; i++) { v[i] = xr[threadIdx.x + 256 * i]; s += v[i]; }
  float mean = blk_sum(s, sm) * (1.0f / Dm);
  float vs = 0.f;
  #pragma unroll
  for (int i = 0; i < 4; i++) { float d = v[i] - mean; vs += d * d; }
  float var = blk_sum(vs, sm) * (1.0f / Dm);
  float inv = rsqrtf(var + 1e-5f);
  _Float16* orow = out + (size_t)row * Dm;
  #pragma unroll
  for (int i = 0; i < 4; i++) {
    int c = threadIdx.x + 256 * i;
    orow[c] = (_Float16)((v[i] - mean) * inv * g[c] + bta[c]);
  }
}

// ---------------- WMMA GEMM: C[M,N] = A[M,K] * W[N,K]^T + bias ----------------
// Double-buffered async-to-LDS staging of 128x64 A and B tiles.
// EPI: 0 = store f16, 1 = qkv scatter (q scaled, v transposed),
//      2 = f32 + residual, 3 = gelu -> f16, 4 = gelu + residual -> f32
__device__ __forceinline__ float gelu_erf(float x) {
  return 0.5f * x * (1.0f + erff(x * 0.70710678118654752f));
}

template <int EPI>
__global__ __launch_bounds__(256) void gemm16_kernel(
    const _Float16* __restrict__ A, const _Float16* __restrict__ W,
    const float* __restrict__ bias, int M, int Nn, int K,
    _Float16* __restrict__ out16, float* __restrict__ out32,
    const float* __restrict__ res,
    _Float16* __restrict__ qb, _Float16* __restrict__ kbuf, _Float16* __restrict__ vtb) {
  __shared__ _Float16 sA[2][128 * KSTEP];   // 2 x 16 KB
  __shared__ _Float16 sB[2][128 * KSTEP];   // 2 x 16 KB
  const int t    = threadIdx.x;
  const int lane = t & 31;
  const int wave = t >> 5;
  const int half = lane >> 4;
  const int lm   = lane & 15;
  const int wm = wave & 3;   // 4 waves along M
  const int wn = wave >> 2;  // 2 waves along N
  const int mblk = blockIdx.y * 128;
  const int nblk = blockIdx.x * 128;
  const int m0 = mblk + wm * 32;
  const int n0 = nblk + wn * 64;

  v8f acc[2][4];
  #pragma unroll
  for (int i = 0; i < 2; i++)
    #pragma unroll
    for (int j = 0; j < 4; j++) acc[i][j] = (v8f){};

  // stage one 128 x KSTEP tile of A and B into LDS buffer `buf` (async DMA).
  // 128 rows * 128B/row = 1024 16B-chunks per matrix; 256 threads * 4 chunks.
  auto stage = [&](int k0, int buf) {
    #pragma unroll
    for (int i = 0; i < 4; i++) {
      int cid = t + 256 * i;
      int row = cid >> 3;          // KSTEP*2/16 = 8 chunks per row
      int cc  = cid & 7;
      const _Float16* ga = A + (size_t)(mblk + row) * K + k0 + cc * 8;
      async_ld_b128((unsigned)(uintptr_t)&sA[buf][row * KSTEP + cc * 8], ga);
      const _Float16* gw = W + (size_t)(nblk + row) * K + k0 + cc * 8;
      async_ld_b128((unsigned)(uintptr_t)&sB[buf][row * KSTEP + cc * 8], gw);
    }
  };

  stage(0, 0);
  int buf = 0;
  for (int k0 = 0; k0 < K; k0 += KSTEP) {
    wait_async0();        // my DMA into buf complete
    __syncthreads();      // everyone's DMA complete; prior-stage reads all done
    if (k0 + KSTEP < K) stage(k0 + KSTEP, buf ^ 1);

    const _Float16* la0 = &sA[buf][(wm * 32 + lm) * KSTEP];
    const _Float16* la1 = &sA[buf][(wm * 32 + 16 + lm) * KSTEP];
    #pragma unroll
    for (int ks = 0; ks < KSTEP; ks += 32) {
      v16h a0 = ld2x8(la0 + ks + 8 * half, la0 + ks + 16 + 8 * half);
      v16h a1 = ld2x8(la1 + ks + 8 * half, la1 + ks + 16 + 8 * half);
      #pragma unroll
      for (int nt = 0; nt < 4; nt++) {
        const _Float16* lb = &sB[buf][(wn * 64 + nt * 16 + lm) * KSTEP + ks + 8 * half];
        v16h bf = ld2x8(lb, lb + 16);
        acc[0][nt] = wmma_f16(a0, bf, acc[0][nt]);
        acc[1][nt] = wmma_f16(a1, bf, acc[1][nt]);
      }
    }
    buf ^= 1;
  }

  #pragma unroll
  for (int mt = 0; mt < 2; mt++) {
    #pragma unroll
    for (int nt = 0; nt < 4; nt++) {
      int n = n0 + nt * 16 + lm;
      float bv = bias[n];
      #pragma unroll
      for (int r = 0; r < 8; r++) {
        int m = m0 + mt * 16 + r + 8 * half;
        float val = acc[mt][nt][r] + bv;
        if constexpr (EPI == 0) {
          out16[(size_t)m * Nn + n] = (_Float16)val;
        } else if constexpr (EPI == 1) {
          int b = m >> 10, tk = m & (Ntok - 1);
          int sel = n >> 10, d = n & (Dm - 1);
          int h = d >> 6, dd = d & 63;
          size_t bh = (size_t)b * Hh + h;
          if (sel == 0)       qb  [(bh * Ntok + tk) * HDd + dd] = (_Float16)(val * 0.125f);
          else if (sel == 1)  kbuf[(bh * Ntok + tk) * HDd + dd] = (_Float16)val;
          else                vtb [(bh * HDd + dd) * Ntok + tk] = (_Float16)val;
        } else if constexpr (EPI == 2) {
          out32[(size_t)m * Nn + n] = val + res[(size_t)m * Nn + n];
        } else if constexpr (EPI == 3) {
          out16[(size_t)m * Nn + n] = (_Float16)gelu_erf(val);
        } else {
          out32[(size_t)m * Nn + n] = gelu_erf(val) + res[(size_t)m * Nn + n];
        }
      }
    }
  }
}

// ---------------- flash attention: one wave per 16 q-rows, kk-chunks of 64 ----------------
// Q,K: [B,H,N,HD] f16 (q pre-scaled by 1/sqrt(HD)); Vt: [B,H,HD,N] f16
// ctx out: [B*N, D] f16 with col = h*HD + d
__global__ __launch_bounds__(256) void attn_kernel(const _Float16* __restrict__ Q,
                                                   const _Float16* __restrict__ Kt,
                                                   const _Float16* __restrict__ Vt,
                                                   _Float16* __restrict__ ctx) {
  __shared__ _Float16 lds[8][16 * 64];   // per-wave 2 KB P tile
  const int lane = threadIdx.x & 31;
  const int wave = threadIdx.x >> 5;
  const int half = lane >> 4;
  const int lm   = lane & 15;

  const int blocksPerBH = Ntok / 128;                 // 8
  const int bh = blockIdx.x / blocksPerBH;
  const int q0 = (blockIdx.x % blocksPerBH) * 128 + wave * 16;
  const int b = bh >> 4, h = bh & 15;

  const _Float16* qrow = Q  + ((size_t)bh * Ntok + q0) * HDd + (size_t)lm * HDd;
  const _Float16* kb   = Kt + (size_t)bh * Ntok * HDd;
  const _Float16* vb   = Vt + (size_t)bh * HDd * Ntok;

  v16h aq0 = ld2x8(qrow + 8 * half,      qrow + 16 + 8 * half);   // d 0..31
  v16h aq1 = ld2x8(qrow + 32 + 8 * half, qrow + 48 + 8 * half);   // d 32..63

  v8f O[4];
  #pragma unroll
  for (int i = 0; i < 4; i++) O[i] = (v8f){};
  float mrow[8], lrow[8];
  #pragma unroll
  for (int r = 0; r < 8; r++) { mrow[r] = -1e30f; lrow[r] = 0.f; }

  _Float16* pl = &lds[wave][0];

  for (int kk = 0; kk < Ntok; kk += 64) {
    v8f S[4];
    #pragma unroll
    for (int i = 0; i < 4; i++) S[i] = (v8f){};
    #pragma unroll
    for (int kt = 0; kt < 4; kt++) {   // 16 k-rows per tile
      const _Float16* kr = kb + (size_t)(kk + 16 * kt + lm) * HDd;
      v16h b0 = ld2x8(kr + 8 * half,      kr + 16 + 8 * half);        // d 0..31
      v16h b1 = ld2x8(kr + 32 + 8 * half, kr + 48 + 8 * half);   // d 32..63
      S[kt] = wmma_f16(aq0, b0, S[kt]);
      S[kt] = wmma_f16(aq1, b1, S[kt]);
    }

    float corr[8];
    #pragma unroll
    for (int r = 0; r < 8; r++) {
      float mx = fmaxf(fmaxf(S[0][r], S[1][r]), fmaxf(S[2][r], S[3][r]));
      mx = fmaxf(mx, __shfl_xor(mx, 1));
      mx = fmaxf(mx, __shfl_xor(mx, 2));
      mx = fmaxf(mx, __shfl_xor(mx, 4));
      mx = fmaxf(mx, __shfl_xor(mx, 8));
      float mn = fmaxf(mrow[r], mx);
      corr[r] = __expf(mrow[r] - mn);
      mrow[r] = mn;
      float p0 = __expf(S[0][r] - mn);
      float p1 = __expf(S[1][r] - mn);
      float p2 = __expf(S[2][r] - mn);
      float p3 = __expf(S[3][r] - mn);
      S[0][r] = p0; S[1][r] = p1; S[2][r] = p2; S[3][r] = p3;
      float sum = (p0 + p1) + (p2 + p3);
      sum += __shfl_xor(sum, 1);
      sum += __shfl_xor(sum, 2);
      sum += __shfl_xor(sum, 4);
      sum += __shfl_xor(sum, 8);
      lrow[r] = lrow[r] * corr[r] + sum;
    }
    #pragma unroll
    for (int td = 0; td < 4; td++)
      #pragma unroll
      for (int r = 0; r < 8; r++) O[td][r] *= corr[r];

    // P (C-layout) -> LDS -> two A-fragments. Per-wave region; same-wave LDS
    // ops stay in order.
    #pragma unroll
    for (int r = 0; r < 8; r++) {
      int row = r + 8 * half;
      pl[row * 64 + lm]      = (_Float16)S[0][r];
      pl[row * 64 + 16 + lm] = (_Float16)S[1][r];
      pl[row * 64 + 32 + lm] = (_Float16)S[2][r];
      pl[row * 64 + 48 + lm] = (_Float16)S[3][r];
    }
    v16h ap0 = ld2x8(pl + lm * 64 + 8 * half,      pl + lm * 64 + 16 + 8 * half); // kk 0..31
    v16h ap1 = ld2x8(pl + lm * 64 + 32 + 8 * half, pl + lm * 64 + 48 + 8 * half); // kk 32..63

    #pragma unroll
    for (int td = 0; td < 4; td++) {
      const _Float16* vr = vb + (size_t)(16 * td + lm) * Ntok + kk + 8 * half;
      v16h bv0 = ld2x8(vr, vr + 16);
      v16h bv1 = ld2x8(vr + 32, vr + 48);
      O[td] = wmma_f16(ap0, bv0, O[td]);
      O[td] = wmma_f16(ap1, bv1, O[td]);
    }
  }

  #pragma unroll
  for (int td = 0; td < 4; td++) {
    #pragma unroll
    for (int r = 0; r < 8; r++) {
      int tok = q0 + r + 8 * half;
      int col = h * HDd + 16 * td + lm;
      ctx[((size_t)b * Ntok + tok) * Dm + col] = (_Float16)(O[td][r] / lrow[r]);
    }
  }
}

// ---------------- host orchestration ----------------
extern "C" void kernel_launch(void* const* d_in, const int* in_sizes, int n_in,
                              void* d_out, int out_size, void* d_ws, size_t ws_size,
                              hipStream_t stream) {
  const float* x      = (const float*)d_in[0];
  const float* ln1_g  = (const float*)d_in[1];
  const float* ln1_b  = (const float*)d_in[2];
  const float* ln2_g  = (const float*)d_in[3];
  const float* ln2_b  = (const float*)d_in[4];
  const float* W_qkv  = (const float*)d_in[5];
  const float* b_qkv  = (const float*)d_in[6];
  const float* W_proj = (const float*)d_in[7];
  const float* b_proj = (const float*)d_in[8];
  const float* W1     = (const float*)d_in[9];
  const float* b1     = (const float*)d_in[10];
  const float* W2     = (const float*)d_in[11];
  const float* b2     = (const float*)d_in[12];

  char* ws = (char*)d_ws;
  _Float16* Wqkv16  = (_Float16*)(ws + 0);            //  6 MB  [3072,1024]
  _Float16* Wproj16 = (_Float16*)(ws + 6291456);      //  2 MB  [1024,1024]
  _Float16* W116    = (_Float16*)(ws + 8388608);      //  8 MB  [4096,1024]
  _Float16* W216    = (_Float16*)(ws + 16777216);     //  8 MB  [1024,4096]
  _Float16* h16     = (_Float16*)(ws + 25165824);     // 16 MB  [8192,1024] (ln1 then ln2)
  _Float16* qb      = (_Float16*)(ws + 41943040);     // 16 MB  [B,H,N,HD]
  _Float16* kbuf    = (_Float16*)(ws + 58720256);     // 16 MB  [B,H,N,HD]
  _Float16* vtb     = (_Float16*)(ws + 75497472);     // 16 MB  [B,H,HD,N]
  _Float16* ctx16   = (_Float16*)(ws + 92274688);     // 16 MB  [8192,1024]
  float*    out1    = (float*)   (ws + 109051904);    // 32 MB  [8192,1024] fp32
  _Float16* ff1     = (_Float16*)(ws + 41943040);     // 64 MB  [8192,4096], reuses dead q/k/vT/ctx

  // 1) weights fp32 -> f16
  cvt_f16_kernel<<<(3 * Dm * Dm + 255) / 256, 256, 0, stream>>>(W_qkv, Wqkv16, 3 * Dm * Dm);
  cvt_f16_kernel<<<(Dm * Dm + 255) / 256, 256, 0, stream>>>(W_proj, Wproj16, Dm * Dm);
  cvt_f16_kernel<<<(FFd * Dm + 255) / 256, 256, 0, stream>>>(W1, W116, FFd * Dm);
  cvt_f16_kernel<<<(Dm * FFd + 255) / 256, 256, 0, stream>>>(W2, W216, Dm * FFd);

  // 2) ln1
  ln_kernel<<<Mtot, 256, 0, stream>>>(x, ln1_g, ln1_b, h16);

  // 3) qkv = h @ Wqkv^T + b  (scatter into q (pre-scaled), k, v^T)
  gemm16_kernel<1><<<dim3(3 * Dm / 128, Mtot / 128), 256, 0, stream>>>(
      h16, Wqkv16, b_qkv, Mtot, 3 * Dm, Dm, nullptr, nullptr, nullptr, qb, kbuf, vtb);

  // 4) flash attention -> ctx [8192,1024] f16
  attn_kernel<<<Bb * Hh * (Ntok / 128), 256, 0, stream>>>(qb, kbuf, vtb, ctx16);

  // 5) out1 = ctx @ Wproj^T + b + x   (fp32 residual)
  gemm16_kernel<2><<<dim3(Dm / 128, Mtot / 128), 256, 0, stream>>>(
      ctx16, Wproj16, b_proj, Mtot, Dm, Dm, nullptr, out1, x, nullptr, nullptr, nullptr);

  // 6) ln2
  ln_kernel<<<Mtot, 256, 0, stream>>>(out1, ln2_g, ln2_b, h16);

  // 7) ff1 = gelu(h2 @ W1^T + b1) -> f16
  gemm16_kernel<3><<<dim3(FFd / 128, Mtot / 128), 256, 0, stream>>>(
      h16, W116, b1, Mtot, FFd, Dm, ff1, nullptr, nullptr, nullptr, nullptr, nullptr);

  // 8) out = gelu(ff1 @ W2^T + b2) + out1 -> d_out fp32
  gemm16_kernel<4><<<dim3(Dm / 128, Mtot / 128), 256, 0, stream>>>(
      ff1, W216, b2, Mtot, Dm, FFd, nullptr, (float*)d_out, out1, nullptr, nullptr, nullptr);

  (void)in_sizes; (void)n_in; (void)out_size; (void)ws_size;
}
